// SparseMoELayer_29008209117691
// MI455X (gfx1250) — matrix operand order repaired and checked by reading, not compiled, not run on hardware
//
#include <hip/hip_runtime.h>
#include <math.h>

// Problem constants (from reference): B=8, S=2048, D=512, E=8, K=2, H=2048
#define T_TOK   16384      // B*S tokens
#define D_DIM   512
#define E_EXP   8
#define H_DIM   2048
#define TILE_M  32
#define MAX_TILES (T_TOK / TILE_M)   // 512 tiles per expert worst-case
#define HCHUNK  256                  // h-chunk width for fused GEMM1->GEMM2

typedef __attribute__((ext_vector_type(16))) __bf16 v16bf;
typedef __attribute__((ext_vector_type(8)))  float  v8f;

union Frag { v16bf v; uint4 q[2]; };

__device__ __forceinline__ unsigned short f2bf(float f) {
  union { float f; unsigned u; } c; c.f = f;
  unsigned u = c.u;
  u += 0x7FFFu + ((u >> 16) & 1u);           // round-to-nearest-even
  return (unsigned short)(u >> 16);
}

__device__ __forceinline__ v8f wmma_bf16(const Frag& a, const Frag& b, v8f c) {
  // D = A(16x32 bf16) x B(32x16 bf16) + C(16x16 f32)
  return __builtin_amdgcn_wmma_f32_16x16x32_bf16(
      false, a.v, false, b.v, (short)0, c, false, false);
}

__device__ __forceinline__ void loadB(Frag& f, const unsigned short* p) {
  const uint4* q = (const uint4*)p;          // K-chunks [p, p+8) and [p+16, p+24) per lane
  f.q[0] = q[0]; f.q[1] = q[2];
}
__device__ __forceinline__ void loadA(Frag& f, const unsigned short* row, int c0) {
  // A layout: per-lane row m; K-chunks [c0, c0+8) and [c0+16, c0+24)
  f.q[0] = *(const uint4*)(row + c0);
  f.q[1] = *(const uint4*)(row + c0 + 16);
}

// ---------------------------------------------------------------------------
// Kernel 1: f32 -> bf16 bulk conversion (8 elems / thread)
// ---------------------------------------------------------------------------
__global__ void cvt_bf16_kernel(const float* __restrict__ src,
                                unsigned short* __restrict__ dst, int n8) {
  int i = blockIdx.x * blockDim.x + threadIdx.x;
  if (i >= n8) return;
  float4 a = ((const float4*)src)[2 * i];
  float4 b = ((const float4*)src)[2 * i + 1];
  uint4 o;
  o.x = (unsigned)f2bf(a.x) | ((unsigned)f2bf(a.y) << 16);
  o.y = (unsigned)f2bf(a.z) | ((unsigned)f2bf(a.w) << 16);
  o.z = (unsigned)f2bf(b.x) | ((unsigned)f2bf(b.y) << 16);
  o.w = (unsigned)f2bf(b.z) | ((unsigned)f2bf(b.w) << 16);
  ((uint4*)dst)[i] = o;
}

// ---------------------------------------------------------------------------
// Kernel 2: gating (one wave32 per token) + bf16 copy of x
// ---------------------------------------------------------------------------
__global__ void gate_kernel(const float* __restrict__ x,
                            const float* __restrict__ gw,
                            unsigned short* __restrict__ xb,
                            int* __restrict__ cnt,
                            int* __restrict__ tok,
                            float* __restrict__ twt) {
  const int wave = threadIdx.x >> 5;
  const int lane = threadIdx.x & 31;
  const int t = blockIdx.x * 8 + wave;

  // each lane owns 16 consecutive f32 of x[t]
  const float4* xr = (const float4*)(x + (size_t)t * D_DIM + lane * 16);
  float4 xv[4];
#pragma unroll
  for (int i = 0; i < 4; ++i) xv[i] = xr[i];

  // emit bf16 x (32B per lane, contiguous)
  {
    uint4 o0, o1;
    o0.x = (unsigned)f2bf(xv[0].x) | ((unsigned)f2bf(xv[0].y) << 16);
    o0.y = (unsigned)f2bf(xv[0].z) | ((unsigned)f2bf(xv[0].w) << 16);
    o0.z = (unsigned)f2bf(xv[1].x) | ((unsigned)f2bf(xv[1].y) << 16);
    o0.w = (unsigned)f2bf(xv[1].z) | ((unsigned)f2bf(xv[1].w) << 16);
    o1.x = (unsigned)f2bf(xv[2].x) | ((unsigned)f2bf(xv[2].y) << 16);
    o1.y = (unsigned)f2bf(xv[2].z) | ((unsigned)f2bf(xv[2].w) << 16);
    o1.z = (unsigned)f2bf(xv[3].x) | ((unsigned)f2bf(xv[3].y) << 16);
    o1.w = (unsigned)f2bf(xv[3].z) | ((unsigned)f2bf(xv[3].w) << 16);
    uint4* xo = (uint4*)(xb + (size_t)t * D_DIM + lane * 16);
    xo[0] = o0; xo[1] = o1;
  }

  // logits for all 8 experts
  float s[E_EXP];
#pragma unroll
  for (int e = 0; e < E_EXP; ++e) {
    const float4* g = (const float4*)(gw + (size_t)e * D_DIM + lane * 16);
    float acc = 0.f;
#pragma unroll
    for (int i = 0; i < 4; ++i) {
      float4 gv = g[i];
      acc += xv[i].x * gv.x + xv[i].y * gv.y + xv[i].z * gv.z + xv[i].w * gv.w;
    }
    s[e] = acc;
  }
  // butterfly reduce across the wave (wave32)
#pragma unroll
  for (int e = 0; e < E_EXP; ++e)
#pragma unroll
    for (int off = 16; off > 0; off >>= 1)
      s[e] += __shfl_xor(s[e], off, 32);

  if (lane == 0) {
    int i0 = 0; float v0 = s[0];
#pragma unroll
    for (int e = 1; e < E_EXP; ++e)
      if (s[e] > v0) { v0 = s[e]; i0 = e; }       // first index wins ties
    int i1 = -1; float v1 = -3.4e38f;
#pragma unroll
    for (int e = 0; e < E_EXP; ++e)
      if (e != i0 && s[e] > v1) { v1 = s[e]; i1 = e; }
    float ex = __expf(v1 - v0);                    // v1 <= v0, no overflow
    float w0 = 1.f / (1.f + ex);
    float w1 = ex / (1.f + ex);
    int p0 = atomicAdd(&cnt[i0], 1);
    tok[i0 * T_TOK + p0] = t;  twt[i0 * T_TOK + p0] = w0;
    int p1 = atomicAdd(&cnt[i1], 1);
    tok[i1 * T_TOK + p1] = t;  twt[i1 * T_TOK + p1] = w1;
  }
}

// ---------------------------------------------------------------------------
// Kernel 3: fused expert FFN.  block = 256 thr (8 waves), tile = 32 tokens.
//   h = gelu(X @ W1^T + b1);  out += wgt * (h @ W2^T + b2)
// GEMM1/GEMM2 fused per 256-wide h-chunk through LDS.  B operands are
// double-buffered in VGPRs so each k-step's loads are issued one step ahead.
// ---------------------------------------------------------------------------
__global__ void __launch_bounds__(256)
moe_ffn_kernel(const unsigned short* __restrict__ xb,   // [T,D] bf16
               const unsigned short* __restrict__ w1b,  // [E,H,D] bf16
               const unsigned short* __restrict__ w2b,  // [E,D,H] bf16
               const float* __restrict__ b1g,           // [E,H]
               const float* __restrict__ b2g,           // [E,D]
               const int* __restrict__ cnt,
               const int* __restrict__ tok,
               const float* __restrict__ twt,
               float* __restrict__ out) {               // [T,D]
  __shared__ __align__(16) unsigned short xs[TILE_M][D_DIM];   // 32 KB
  __shared__ __align__(16) unsigned short hs[TILE_M][HCHUNK];  // 16 KB
  __shared__ int   s_tok[TILE_M];
  __shared__ float s_wt[TILE_M];

  const int e    = blockIdx.x >> 9;       // / MAX_TILES
  const int tile = blockIdx.x & (MAX_TILES - 1);
  const int n    = cnt[e];
  if (tile * TILE_M >= n) return;

  if (threadIdx.x < TILE_M) {
    int idx   = tile * TILE_M + threadIdx.x;
    int valid = idx < n;
    int src   = valid ? idx : (n - 1);
    s_tok[threadIdx.x] = tok[e * T_TOK + src];
    s_wt[threadIdx.x]  = valid ? twt[e * T_TOK + src] : 0.f;
  }
  __syncthreads();

  // gather X tile into LDS (each thread: 64B of one row)
  {
    int row = threadIdx.x >> 3, seg = threadIdx.x & 7;
    const uint4* src = (const uint4*)(xb + (size_t)s_tok[row] * D_DIM + seg * 64);
    uint4* dst = (uint4*)(&xs[row][seg * 64]);
#pragma unroll
    for (int i = 0; i < 4; ++i) dst[i] = src[i];
  }
  __syncthreads();

  const int wave   = threadIdx.x >> 5;
  const int lane   = threadIdx.x & 31;
  const int laneLo = lane & 15;
  const int laneHi = lane >> 4;

  // per-lane base pointers for the two B operands (ISA B layout: lane = col)
  const unsigned short* w2row[4];
#pragma unroll
  for (int g = 0; g < 4; ++g)
    w2row[g] = w2b + ((size_t)e * D_DIM + wave * 64 + g * 16 + laneLo) * H_DIM
             + laneHi * 8;

  // GEMM2 accumulators: out tile [32, 64] per wave -> 2 Mtiles x 4 Ntiles
  v8f acc2[2][4];
#pragma unroll
  for (int mt = 0; mt < 2; ++mt)
#pragma unroll
    for (int g = 0; g < 4; ++g) acc2[mt][g] = (v8f)(0.f);

  // GEMM2 B prologue (chunk 0, k=0); tail prefetch of each chunk feeds the next
  Frag w2cur[4], w2nxt[4];
#pragma unroll
  for (int g = 0; g < 4; ++g) loadB(w2cur[g], w2row[g]);

  for (int c = 0; c < H_DIM / HCHUNK; ++c) {
    // ---- GEMM1: this wave computes h columns [c*256 + wave*32, +32) ----
    v8f acc1[2][2];
#pragma unroll
    for (int mt = 0; mt < 2; ++mt)
#pragma unroll
      for (int g = 0; g < 2; ++g) acc1[mt][g] = (v8f)(0.f);

    const unsigned short* w1row0 =
        w1b + ((size_t)e * H_DIM + c * HCHUNK + wave * 32 + laneLo) * D_DIM
            + laneHi * 8;
    const unsigned short* w1row1 = w1row0 + (size_t)16 * D_DIM;

    Frag bc0, bc1, bn0, bn1, a0, a1;
    loadB(bc0, w1row0);                     // k = 0
    loadB(bc1, w1row1);
    for (int k0 = 0; k0 < D_DIM; k0 += 64) {
      // step k0: consume bc*, prefetch k0+32 into bn*
      loadB(bn0, w1row0 + k0 + 32);
      loadB(bn1, w1row1 + k0 + 32);
      loadA(a0, &xs[laneLo][0],      k0 + laneHi * 8);
      loadA(a1, &xs[16 + laneLo][0], k0 + laneHi * 8);
      acc1[0][0] = wmma_bf16(a0, bc0, acc1[0][0]);
      acc1[0][1] = wmma_bf16(a0, bc1, acc1[0][1]);
      acc1[1][0] = wmma_bf16(a1, bc0, acc1[1][0]);
      acc1[1][1] = wmma_bf16(a1, bc1, acc1[1][1]);
      // step k0+32: consume bn*, prefetch k0+64 into bc* (tail overrun stays
      // inside the bf16 weight arrays in the workspace; value unused)
      loadB(bc0, w1row0 + k0 + 64);
      loadB(bc1, w1row1 + k0 + 64);
      loadA(a0, &xs[laneLo][0],      k0 + 32 + laneHi * 8);
      loadA(a1, &xs[16 + laneLo][0], k0 + 32 + laneHi * 8);
      acc1[0][0] = wmma_bf16(a0, bn0, acc1[0][0]);
      acc1[0][1] = wmma_bf16(a0, bn1, acc1[0][1]);
      acc1[1][0] = wmma_bf16(a1, bn0, acc1[1][0]);
      acc1[1][1] = wmma_bf16(a1, bn1, acc1[1][1]);
    }
    // bias + exact GELU + pack bf16 into LDS h-tile
#pragma unroll
    for (int mt = 0; mt < 2; ++mt)
#pragma unroll
      for (int g = 0; g < 2; ++g) {
        int col  = wave * 32 + g * 16 + laneLo;              // within chunk
        float bi = b1g[e * H_DIM + c * HCHUNK + col];
#pragma unroll
        for (int r = 0; r < 8; ++r) {
          int m = mt * 16 + laneHi * 8 + r;                  // C layout row
          float v = acc1[mt][g][r] + bi;
          v = 0.5f * v * (1.0f + erff(v * 0.70710678118654752f));
          hs[m][col] = f2bf(v);
        }
      }
    __syncthreads();

    // ---- GEMM2 partial: d-range [wave*64, +64), K = this 256-chunk ----
    const int kbase = c * HCHUNK;
    for (int k0 = 0; k0 < HCHUNK; k0 += 64) {
      // step k0: consume w2cur, prefetch k0+32
#pragma unroll
      for (int g = 0; g < 4; ++g) loadB(w2nxt[g], w2row[g] + kbase + k0 + 32);
      loadA(a0, &hs[laneLo][0],      k0 + laneHi * 8);
      loadA(a1, &hs[16 + laneLo][0], k0 + laneHi * 8);
#pragma unroll
      for (int g = 0; g < 4; ++g) {
        acc2[0][g] = wmma_bf16(a0, w2cur[g], acc2[0][g]);
        acc2[1][g] = wmma_bf16(a1, w2cur[g], acc2[1][g]);
      }
      // step k0+32: consume w2nxt, prefetch k0+64 (last one = next chunk's k0)
#pragma unroll
      for (int g = 0; g < 4; ++g) loadB(w2cur[g], w2row[g] + kbase + k0 + 64);
      loadA(a0, &hs[laneLo][0],      k0 + 32 + laneHi * 8);
      loadA(a1, &hs[16 + laneLo][0], k0 + 32 + laneHi * 8);
#pragma unroll
      for (int g = 0; g < 4; ++g) {
        acc2[0][g] = wmma_bf16(a0, w2nxt[g], acc2[0][g]);
        acc2[1][g] = wmma_bf16(a1, w2nxt[g], acc2[1][g]);
      }
    }
    __syncthreads();   // hs reused next chunk
  }

  // epilogue: +b2, scale by combine weight, accumulate into out
#pragma unroll
  for (int mt = 0; mt < 2; ++mt)
#pragma unroll
    for (int g = 0; g < 4; ++g) {
      int d = wave * 64 + g * 16 + laneLo;
      float bi = b2g[e * D_DIM + d];
#pragma unroll
      for (int r = 0; r < 8; ++r) {
        int m = mt * 16 + laneHi * 8 + r;
        float val = (acc2[mt][g][r] + bi) * s_wt[m];
        float* dst = out + (size_t)s_tok[m] * D_DIM + d;
#if defined(__HIP_DEVICE_COMPILE__)
        unsafeAtomicAdd(dst, val);   // global_atomic_add_f32
#else
        atomicAdd(dst, val);
#endif
      }
    }
}

// ---------------------------------------------------------------------------
extern "C" void kernel_launch(void* const* d_in, const int* in_sizes, int n_in,
                              void* d_out, int out_size, void* d_ws, size_t ws_size,
                              hipStream_t stream) {
  const float* x  = (const float*)d_in[0];   // [B,S,D]
  const float* gw = (const float*)d_in[1];   // [E,D]
  const float* w1 = (const float*)d_in[2];   // [E,H,D]
  const float* b1 = (const float*)d_in[3];   // [E,H]
  const float* w2 = (const float*)d_in[4];   // [E,D,H]
  const float* b2 = (const float*)d_in[5];   // [E,D]
  float* out = (float*)d_out;

  const size_t XB_BYTES = (size_t)T_TOK * D_DIM * 2;
  const size_t W_BYTES  = (size_t)E_EXP * H_DIM * D_DIM * 2;

  char* ws = (char*)d_ws;
  unsigned short* xb  = (unsigned short*)ws;  ws += XB_BYTES;
  unsigned short* w1b = (unsigned short*)ws;  ws += W_BYTES;
  unsigned short* w2b = (unsigned short*)ws;  ws += W_BYTES;
  int*   cnt = (int*)ws;                      ws += 256;
  int*   tokl = (int*)ws;                     ws += (size_t)E_EXP * T_TOK * sizeof(int);
  float* twt = (float*)ws;                    ws += (size_t)E_EXP * T_TOK * sizeof(float);

  hipMemsetAsync(cnt, 0, 256, stream);
  hipMemsetAsync(d_out, 0, (size_t)out_size * sizeof(float), stream);

  const int n8 = (int)((size_t)E_EXP * H_DIM * D_DIM / 8);   // 1M elems*8 each
  cvt_bf16_kernel<<<(n8 + 255) / 256, 256, 0, stream>>>(w1, w1b, n8);
  cvt_bf16_kernel<<<(n8 + 255) / 256, 256, 0, stream>>>(w2, w2b, n8);

  gate_kernel<<<T_TOK / 8, 256, 0, stream>>>(x, gw, xb, cnt, tokl, twt);

  moe_ffn_kernel<<<E_EXP * MAX_TILES, 256, 0, stream>>>(
      xb, w1b, w2b, b1, b2, cnt, tokl, twt, out);
}
